// Decoder_13597866459207
// MI455X (gfx1250) — matrix operand order repaired and checked by reading, not compiled, not run on hardware
//
#include <hip/hip_runtime.h>

// Problem constants (match reference)
#define V_  10000
#define E_  512
#define H_  512
#define B_  64
#define TO_ 128
#define TI_ 1000

#define NTW 4   // N-tiles (16 cols each) per wave
#define MT_ 4   // M-tiles (waves) per block = 64 rows / 16

typedef __attribute__((ext_vector_type(16))) __bf16        v16bf;
typedef __attribute__((ext_vector_type(8)))  float         v8f;
typedef __attribute__((ext_vector_type(4)))  float         v4f;
typedef __attribute__((ext_vector_type(4)))  unsigned int  v4u;

struct Frag32B { v4u lo, hi; };

__device__ __forceinline__ unsigned short f2bf(float f) {
  unsigned int u = __builtin_bit_cast(unsigned int, f);
  u += 0x7FFFu + ((u >> 16) & 1u);            // round-to-nearest-even
  return (unsigned short)(u >> 16);
}

// A fragment: row-major bf16 A, per-lane 8 contiguous dwords at p[0..7]
__device__ __forceinline__ v16bf ldfragA(const unsigned int* p) {
  Frag32B t;
  t.lo = *(const v4u*)p;
  t.hi = *(const v4u*)(p + 4);
  return __builtin_bit_cast(v16bf, t);
}
// B fragment: tiled-packed layout, 8 contiguous dwords per lane
__device__ __forceinline__ v16bf ldfragB(const unsigned int* p) {
  Frag32B t;
  t.lo = *(const v4u*)p;
  t.hi = *(const v4u*)(p + 4);
  return __builtin_bit_cast(v16bf, t);
}

// ---------------------------------------------------------------------------
// Pack row-major (N x K) f32 weight into fragment-native tiled bf16 layout:
//   out[((nT*KC + kc)*32 + lane)*8 + j] = pack(W[n, k], W[n, k+1])
//   n = nT*16 + (lane&15),  k = kc*32 + (lane>=16)*16 + 2j
// so a wave's B fragment for one 16x32 K-chunk is one contiguous 1KB block.
// ---------------------------------------------------------------------------
__global__ void pack_w_tiled(const float* __restrict__ W, unsigned int* __restrict__ out,
                             int N, int K) {
  int idx = blockIdx.x * blockDim.x + threadIdx.x;
  int KC = K >> 5;
  int total = (N >> 4) * KC * 256;
  if (idx >= total) return;
  int j    = idx & 7;
  int lane = (idx >> 3) & 31;
  int kc   = (idx >> 8) % KC;
  int nT   = (idx >> 8) / KC;
  int n    = nT * 16 + (lane & 15);
  int k    = (kc << 5) + ((lane >> 4) << 4) + (j << 1);
  float f0 = W[(size_t)n * K + k];
  float f1 = W[(size_t)n * K + k + 1];
  out[idx] = (unsigned int)f2bf(f0) | ((unsigned int)f2bf(f1) << 16);
}

// Fused gate weight Wg (2048 x 1536) = [W_ih (2048x1024) | W_hh (2048x512)]
__global__ void pack_wg_tiled(const float* __restrict__ Wih, const float* __restrict__ Whh,
                              unsigned int* __restrict__ out) {
  const int N = 4 * H_, K = E_ + 2 * H_;       // 2048, 1536
  int idx = blockIdx.x * blockDim.x + threadIdx.x;
  int KC = K >> 5;
  int total = (N >> 4) * KC * 256;
  if (idx >= total) return;
  int j    = idx & 7;
  int lane = (idx >> 3) & 31;
  int kc   = (idx >> 8) % KC;
  int nT   = (idx >> 8) / KC;
  int n    = nT * 16 + (lane & 15);
  int k    = (kc << 5) + ((lane >> 4) << 4) + (j << 1);
  float f0, f1;
  if (k < E_ + H_) { f0 = Wih[(size_t)n * (E_ + H_) + k];
                     f1 = Wih[(size_t)n * (E_ + H_) + k + 1]; }
  else             { int kk = k - (E_ + H_);
                     f0 = Whh[(size_t)n * H_ + kk];
                     f1 = Whh[(size_t)n * H_ + kk + 1]; }
  out[idx] = (unsigned int)f2bf(f0) | ((unsigned int)f2bf(f1) << 16);
}

// Embedding gather for all timesteps -> bf16, layout [t][b][e]
__global__ void embed_all(const int* __restrict__ tok, const float* __restrict__ emb,
                          unsigned short* __restrict__ out) {
  int idx = blockIdx.x * blockDim.x + threadIdx.x;
  const int total = TO_ * B_ * E_;
  if (idx >= total) return;
  int e = idx & (E_ - 1);
  int r = idx >> 9;          // / 512
  int b = r & (B_ - 1);
  int t = r >> 6;            // / 64
  int token = tok[b * TO_ + t];
  out[idx] = f2bf(emb[(size_t)token * E_ + e]);
}

// ---------------------------------------------------------------------------
// Split-K bf16 WMMA GEMM partial pass: part[slice][m][n] = A_slice @ W^T.
// Scalar work decomposition: blockIdx.x = slice * nQuads + nQuad (SALU-only),
// so all K-loop bounds are SGPRs -> scalar loop control, EXEC all-ones, and
// pointer-walking induction (constant-offset b128 clauses) in the hot loop.
// Wave id within the 128-thread block = M tile (4 x 16 rows = 64).
// A: bf16 row-major, two column sections (A0: [0,ksplit), A1: [ksplit,K)).
// B: fragment-native tiles from pack_w_tiled, padded to NTW tiles so the
// tail needs no pointer clamp (pad fragments computed, discarded at store).
// ---------------------------------------------------------------------------
__global__ void wgemm_bf16(
    const unsigned int* __restrict__ A0, int lda0,   // row stride in pairs
    const unsigned int* __restrict__ A1, int lda1,
    int ksplit,
    const unsigned int* __restrict__ Bt,             // tiled packed B (padded)
    int N, int K, int KS,
    float* __restrict__ part,                        // KS x 64 x N f32
    int nTiles)
{
  int lane = threadIdx.x & 31;
  int mT   = threadIdx.x >> 5;                 // wave id = M tile index
  int nQuads = (nTiles + NTW - 1) / NTW;
  int slice  = blockIdx.x / nQuads;            // scalar
  int nT0    = (blockIdx.x % nQuads) * NTW;    // scalar

  int KC   = K >> 5;
  int kPer = K / KS;                           // multiple of 32 by construction
  int kbeg = slice * kPer;                     // scalar
  int kend = kbeg + kPer;                      // scalar
  int mrow = mT * 16 + (lane & 15);
  int hi4  = (lane >> 4) << 3;

  const unsigned int* bptr[NTW];
#pragma unroll
  for (int s = 0; s < NTW; ++s)
    bptr[s] = Bt + ((size_t)(nT0 + s) * KC + (kbeg >> 5)) * 256 + lane * 8;

  const unsigned int* arow0 = A0 + (size_t)mrow * lda0 + hi4;
  const unsigned int* arow1 = A1 + (size_t)mrow * lda1 + hi4;

  int kspl  = (ksplit < K) ? ksplit : K;       // all scalar
  int loEnd = (kend < kspl) ? kend : kspl;
  int hiBeg = (kbeg > kspl) ? kbeg : kspl;
  int n0 = (loEnd > kbeg) ? ((loEnd - kbeg) >> 5) : 0;
  int n1 = (kend > hiBeg) ? ((kend - hiBeg) >> 5) : 0;

  v8f acc[NTW] = {};
  const unsigned int* aptr = arow0 + (kbeg >> 1);
  for (int i = 0; i < n0; ++i) {
    v16bf a = ldfragA(aptr); aptr += 16;
    v16bf b[NTW];
#pragma unroll
    for (int s = 0; s < NTW; ++s) { b[s] = ldfragB(bptr[s]); bptr[s] += 256; }
#pragma unroll
    for (int s = 0; s < NTW; ++s)
      acc[s] = __builtin_amdgcn_wmma_f32_16x16x32_bf16(false, a, false, b[s],
                                                       (short)0, acc[s], false, false);
  }
  aptr = arow1 + ((hiBeg - kspl) >> 1);
  for (int i = 0; i < n1; ++i) {
    v16bf a = ldfragA(aptr); aptr += 16;
    v16bf b[NTW];
#pragma unroll
    for (int s = 0; s < NTW; ++s) { b[s] = ldfragB(bptr[s]); bptr[s] += 256; }
#pragma unroll
    for (int s = 0; s < NTW; ++s)
      acc[s] = __builtin_amdgcn_wmma_f32_16x16x32_bf16(false, a, false, b[s],
                                                       (short)0, acc[s], false, false);
  }

  // Store partials: D layout VGPR r -> M = r + 8*(lane>=16), N = lane&15
  int hi8 = (lane >> 4) << 3;
  float* pslice = part + (size_t)slice * 64 * N;
#pragma unroll
  for (int s = 0; s < NTW; ++s) {
    if (nT0 + s >= nTiles) continue;          // scalar store mask (tail only)
    int ncol = (nT0 + s) * 16 + (lane & 15);
#pragma unroll
    for (int r = 0; r < 8; ++r) {
      int m = mT * 16 + hi8 + r;
      pslice[(size_t)m * N + ncol] = acc[s][r];
    }
  }
}

// ---------------------------------------------------------------------------
// Split-K reduction + bias + epilogue.
//  mode 0: f32 store  out[(m*rowScale+rowOff)*ldOut + n] = sum + bias
//  mode 1: bf16 store outBf[m*ldOutBf + n] = bf16(tanh(sum + bias))
// ---------------------------------------------------------------------------
__global__ void reduce_out(const float* __restrict__ part, int KS, int N,
                           const float* __restrict__ bias0,
                           const float* __restrict__ bias1,
                           float* __restrict__ outF, int ldOut, int rowScale, int rowOff,
                           unsigned short* __restrict__ outBf, int ldOutBf, int mode)
{
  int idx = blockIdx.x * blockDim.x + threadIdx.x;
  if (idx >= 64 * N) return;
  int m = idx / N;
  int n = idx - m * N;
  float s = 0.f;
  for (int k = 0; k < KS; ++k) s += part[(size_t)(k * 64 + m) * N + n];
  s += bias0[n];
  if (bias1) s += bias1[n];
  if (mode == 0) {
    outF[(size_t)(m * rowScale + rowOff) * ldOut + n] = s;
  } else {
    outBf[(size_t)m * ldOutBf + n] = f2bf(tanhf(s));
  }
}

// ---------------------------------------------------------------------------
// LSTM cell pointwise update. PyTorch gate order: i,f,g,o along N.
// ---------------------------------------------------------------------------
__global__ void lstm_update(const float* __restrict__ gates,
                            float* __restrict__ h, float* __restrict__ c,
                            unsigned short* __restrict__ rnna2,   // (B,1024)=[att|h]
                            unsigned short* __restrict__ mlpa)    // (B,1024)=[h|att]
{
  int idx = blockIdx.x * blockDim.x + threadIdx.x;
  if (idx >= B_ * H_) return;
  int b = idx >> 9;
  int j = idx & 511;
  const float* g = gates + (size_t)b * (4 * H_);
  float gi = g[j], gf = g[j + H_], gg = g[j + 2 * H_], go = g[j + 3 * H_];
  float si = 1.f / (1.f + __expf(-gi));
  float sf = 1.f / (1.f + __expf(-gf));
  float so = 1.f / (1.f + __expf(-go));
  float cn = sf * c[idx] + si * tanhf(gg);
  float hn = so * tanhf(cn);
  c[idx] = cn;
  h[idx] = hn;
  unsigned short hb = f2bf(hn);
  rnna2[b * 1024 + 512 + j] = hb;
  mlpa[b * 1024 + j] = hb;
}

// ---------------------------------------------------------------------------
// Dot-product attention + softmax + context, one block per batch row.
// enc (131 MB) is L2-resident; scores/softmax live in LDS; float4 vectorized.
// ---------------------------------------------------------------------------
__global__ void attention(const float* __restrict__ enc,
                          const float* __restrict__ h,
                          unsigned short* __restrict__ rnna2,
                          unsigned short* __restrict__ mlpa)
{
  __shared__ __align__(16) float hs[H_];
  __shared__ __align__(16) float sc[TI_];
  __shared__ float red[256];
  int b = blockIdx.x;
  int tid = threadIdx.x;
  const float* eb = enc + (size_t)b * TI_ * H_;

  for (int i = tid; i < H_; i += 256) hs[i] = h[b * H_ + i];
  __syncthreads();

  const v4f* hs4 = (const v4f*)hs;
  for (int t = tid; t < TI_; t += 256) {
    const v4f* er4 = (const v4f*)(eb + (size_t)t * H_);
    float acc = 0.f;
#pragma unroll 4
    for (int k = 0; k < H_ / 4; ++k) {
      v4f e = er4[k], hv = hs4[k];
      acc += e[0]*hv[0] + e[1]*hv[1] + e[2]*hv[2] + e[3]*hv[3];
    }
    sc[t] = acc;
  }
  __syncthreads();

  float mx = -3.4e38f;
  for (int t = tid; t < TI_; t += 256) mx = fmaxf(mx, sc[t]);
  red[tid] = mx; __syncthreads();
  for (int s = 128; s > 0; s >>= 1) {
    if (tid < s) red[tid] = fmaxf(red[tid], red[tid + s]);
    __syncthreads();
  }
  mx = red[0];
  __syncthreads();

  float sum = 0.f;
  for (int t = tid; t < TI_; t += 256) { float v = __expf(sc[t] - mx); sc[t] = v; sum += v; }
  red[tid] = sum; __syncthreads();
  for (int s = 128; s > 0; s >>= 1) {
    if (tid < s) red[tid] += red[tid + s];
    __syncthreads();
  }
  float inv = 1.f / red[0];
  __syncthreads();

  const v4f* sc4 = (const v4f*)sc;
  for (int cidx = tid; cidx < H_; cidx += 256) {
    const float* ec = eb + cidx;
    float acc = 0.f;
#pragma unroll 2
    for (int q = 0; q < TI_ / 4; ++q) {
      v4f w = sc4[q];
      const float* e0 = ec + (size_t)(4 * q) * H_;
      acc += w[0]*e0[0] + w[1]*e0[H_] + w[2]*e0[2*H_] + w[3]*e0[3*H_];
    }
    unsigned short ab = f2bf(acc * inv);
    rnna2[b * 1024 + cidx] = ab;          // att section of rnn_in
    mlpa[b * 1024 + 512 + cidx] = ab;     // att section of mlp_in
  }
}

// ---------------------------------------------------------------------------
extern "C" void kernel_launch(void* const* d_in, const int* in_sizes, int n_in,
                              void* d_out, int out_size, void* d_ws, size_t ws_size,
                              hipStream_t stream) {
  (void)in_sizes; (void)n_in; (void)out_size; (void)ws_size;
  const int*   tok = (const int*)d_in[0];
  const float* enc = (const float*)d_in[1];
  const float* emb = (const float*)d_in[2];
  const float* Wih = (const float*)d_in[3];
  const float* bih = (const float*)d_in[4];
  const float* Whh = (const float*)d_in[5];
  const float* bhh = (const float*)d_in[6];
  const float* W1  = (const float*)d_in[7];
  const float* b1  = (const float*)d_in[8];
  const float* W2  = (const float*)d_in[9];
  const float* b2  = (const float*)d_in[10];
  float* out = (float*)d_out;

  // Packed-B tile counts (padded to NTW multiples; pad read-but-discarded)
  const int wgT = 128, wgKC = 48;              // gates: N=2048, K=1536
  const int w1T = 32,  w1KC = 32;              // mlp1:  N=512,  K=1024
  const int w2T = 628, w2KC = 16;              // logits: 625 -> pad 628, K=512

  // Workspace carve-out (~33 MB total)
  char* ws = (char*)d_ws;
  size_t off = 0;
  auto carve = [&](size_t bytes) -> char* {
    char* p = ws + off; off += (bytes + 255) & ~(size_t)255; return p;
  };
  unsigned int*   WgP   = (unsigned int*)  carve(sizeof(unsigned int) * (size_t)wgT * wgKC * 256);
  unsigned int*   W1P   = (unsigned int*)  carve(sizeof(unsigned int) * (size_t)w1T * w1KC * 256);
  unsigned int*   W2P   = (unsigned int*)  carve(sizeof(unsigned int) * (size_t)w2T * w2KC * 256);
  unsigned short* EMB   = (unsigned short*)carve(sizeof(unsigned short) * (size_t)TO_ * B_ * E_);
  unsigned short* RNNA2 = (unsigned short*)carve(sizeof(unsigned short) * (size_t)B_ * 1024);
  unsigned short* MLPA  = (unsigned short*)carve(sizeof(unsigned short) * (size_t)B_ * 1024);
  unsigned short* HIDA  = (unsigned short*)carve(sizeof(unsigned short) * (size_t)B_ * H_);
  float*          GATES = (float*)         carve(sizeof(float) * (size_t)B_ * 4 * H_);
  float*          Hf    = (float*)         carve(sizeof(float) * (size_t)B_ * H_);
  float*          Cf    = (float*)         carve(sizeof(float) * (size_t)B_ * H_);
  float*          PART  = (float*)         carve(sizeof(float) * (size_t)2 * 64 * V_); // max slices*64*N

  // One-time: pack weights to fragment-native bf16 tiles, gather embeddings
  { int total = 768 * 2048;    pack_wg_tiled<<<(total + 255) / 256, 256, 0, stream>>>(Wih, Whh, WgP); }
  { int total = 512 * 512;     pack_w_tiled <<<(total + 255) / 256, 256, 0, stream>>>(W1, W1P, 512, 1024); }
  { int total = 256 * 10000;   pack_w_tiled <<<(total + 255) / 256, 256, 0, stream>>>(W2, W2P, 10000, 512); }
  { int total = TO_ * B_ * E_; embed_all    <<<(total + 255) / 256, 256, 0, stream>>>(tok, emb, EMB); }

  // Recurrent state init (graph-capture safe, deterministic per call)
  hipMemsetAsync(RNNA2, 0, sizeof(unsigned short) * (size_t)B_ * 1024, stream);
  hipMemsetAsync(MLPA,  0, sizeof(unsigned short) * (size_t)B_ * 1024, stream);
  hipMemsetAsync(Hf,    0, sizeof(float) * (size_t)B_ * H_, stream);
  hipMemsetAsync(Cf,    0, sizeof(float) * (size_t)B_ * H_, stream);

  for (int t = 0; t < TO_; ++t) {
    const unsigned int* Axt = (const unsigned int*)(EMB + (size_t)t * B_ * E_);

    // gates = [x_t | att | h] @ [W_ih | W_hh]^T + b_ih + b_hh (M=64,N=2048,K=1536, KS=4)
    {
      int nT = 128, KS = 4, nQ = (nT + NTW - 1) / NTW;
      wgemm_bf16<<<KS * nQ, 32 * MT_, 0, stream>>>(
          Axt, E_ / 2, (const unsigned int*)RNNA2, 512, /*ksplit=*/512,
          WgP, 2048, 1536, KS, PART, nT);
      int elems = 64 * 2048;
      reduce_out<<<(elems + 255) / 256, 256, 0, stream>>>(
          PART, KS, 2048, bih, bhh, GATES, 2048, 1, 0, nullptr, 0, /*mode=*/0);
    }

    lstm_update<<<(B_ * H_ + 255) / 256, 256, 0, stream>>>(GATES, Hf, Cf, RNNA2, MLPA);

    attention<<<B_, 256, 0, stream>>>(enc, Hf, RNNA2, MLPA);

    // hid = tanh([h|att] @ W1^T + b1)   (M=64,N=512,K=1024, KS=4) -> bf16
    {
      int nT = 32, KS = 4, nQ = (nT + NTW - 1) / NTW;
      wgemm_bf16<<<KS * nQ, 32 * MT_, 0, stream>>>(
          (const unsigned int*)MLPA, 512, (const unsigned int*)MLPA, 512, /*ksplit=*/1024,
          W1P, 512, 1024, KS, PART, nT);
      int elems = 64 * 512;
      reduce_out<<<(elems + 255) / 256, 256, 0, stream>>>(
          PART, KS, 512, b1, nullptr, nullptr, 0, 1, 0, HIDA, 512, /*mode=*/1);
    }

    // logits = hid @ W2^T + b2  (M=64,N=10000,K=512, KS=2) -> out[(b*TO+t)*V + n]
    {
      int nT = 625, KS = 2, nQ = (nT + NTW - 1) / NTW;
      wgemm_bf16<<<KS * nQ, 32 * MT_, 0, stream>>>(
          (const unsigned int*)HIDA, 256, (const unsigned int*)HIDA, 256, /*ksplit=*/512,
          W2P, 10000, 512, KS, PART, nT);
      int elems = 64 * V_;
      reduce_out<<<(elems + 255) / 256, 256, 0, stream>>>(
          PART, KS, V_, b2, nullptr, out, V_, TO_, t, nullptr, 0, /*mode=*/0);
    }
  }
}
